// VQ_Layer_76355928588882
// MI455X (gfx1250) — compile-verified, matrix operand order checked by reading, not run on hardware
//
#include <hip/hip_runtime.h>

// ---------------------------------------------------------------------------
// MI455X (gfx1250) implementation.
// Heavy path: 3x bidirectional GRU (T=96, N=2048, H=128) via
// v_wmma_f32_16x16x32_f16 with the recurrence kept on-chip (h in LDS),
// one workgroup per 16-sample batch tile, weights pre-packed into native
// WMMA B-fragment layout. X_t tiles staged LDS-side by the Tensor Data Mover
// (tensor_load_to_lds, double-buffered, s_wait_tensorcnt). MLP head also WMMA.
// ---------------------------------------------------------------------------

typedef __attribute__((ext_vector_type(8)))  _Float16 v8h;
typedef __attribute__((ext_vector_type(16))) _Float16 v16h;
typedef __attribute__((ext_vector_type(8)))  float    v8f;
typedef __attribute__((ext_vector_type(4)))  unsigned int v4u;
typedef __attribute__((ext_vector_type(8)))  int      v8i;
typedef __attribute__((ext_vector_type(4)))  int      v4i;

union V16 {
  v16h v;
  v8h  h[2];
  _Float16 e[16];
};

__device__ inline v8f vzero() {
  v8f z;
#pragma unroll
  for (int i = 0; i < 8; ++i) z[i] = 0.f;
  return z;
}

__device__ inline v8f wmma_f16(const v16h& a, const v16h& b, const v8f& c) {
  // D = A(16x32 f16) x B(32x16 f16) + C(16x16 f32)
  return __builtin_amdgcn_wmma_f32_16x16x32_f16(
      /*neg_a=*/false, a, /*neg_b=*/false, b,
      /*c_mod=*/(short)0, c, /*reuse_a=*/false, /*reuse_b=*/false);
}

// ---------------------------------------------------------------------------
// Tensor Data Mover: 1-D contiguous tile, global -> LDS.
// D# per CDNA5 ISA:
//   group0: count=1 | lds_addr | global_addr[56:0] | type=2
//   group1: data_size=3 (8B elems), tensor_dim0 large, tensor_dim1=1,
//           tile_dim0 = bytes/8, tile_dim1 = 1, dim0_stride = tile_dim0
// Issued by one wave (TDM ignores EXEC; one instruction per wave = one DMA),
// completion gated with s_wait_tensorcnt + workgroup barrier.
// ---------------------------------------------------------------------------
__device__ inline void tdm_load_1d(unsigned int lds_off, const void* gptr,
                                   unsigned int bytes) {
  const unsigned long long ga = (unsigned long long)(uintptr_t)gptr;
  const unsigned int elems = bytes >> 3;          // 8-byte elements
  v4u g0;
  g0[0] = 1u;                                      // count=1, user mode
  g0[1] = lds_off;                                 // LDS byte address
  g0[2] = (unsigned int)ga;                        // global_addr[31:0]
  g0[3] = (unsigned int)((ga >> 32) & 0x01FFFFFFu) // global_addr[56:32]
          | 0x80000000u;                           // type=2 ("image")
  const unsigned int td0 = 0x100000u;              // tensor_dim0 (no OOB clip)
  v8i g1;
  g1[0] = (int)(3u << 16);                         // wg_mask=0, data_size=8B
  g1[1] = (int)((td0 & 0xFFFFu) << 16);            // tensor_dim0[15:0]
  g1[2] = (int)(((td0 >> 16) & 0xFFFFu)            // tensor_dim0[31:16]
                | (1u << 16));                     // tensor_dim1[15:0] = 1
  g1[3] = (int)(elems << 16);                      // tile_dim0
  g1[4] = 1;                                       // tile_dim1 = 1
  g1[5] = (int)elems;                              // tensor_dim0_stride lo32
  g1[6] = 0;
  g1[7] = 0;
  v4i z4 = {0, 0, 0, 0};
#if __has_include(<hip/amd_detail/amd_gfx1250_TDM.h>)
  v8i z8 = {0, 0, 0, 0, 0, 0, 0, 0};
  __builtin_amdgcn_tensor_load_to_lds(g0, g1, z4, z4, z8, 0);
#else
  __builtin_amdgcn_tensor_load_to_lds(g0, g1, z4, z4, 0);
#endif
}

// (lane,j) -> K index inside a 16-bit 16x32 A / 32x16 B WMMA fragment
__device__ inline int frag_k(int lane, int j) {
  int v = j >> 1;
  return ((lane >> 4) & 1) * 8 + ((v >> 2) & 1) * 16 + (v & 3) * 2 + (j & 1);
}

// ---------------------------------------------------------------------------
// Pack an fp32 row-major weight (rows = nTiles*16 outputs, cols = K) into
// f16 WMMA B-fragments: tile (nt,kt) is 512 halfs, lane l owns halfs
// [l*16, l*16+16) so runtime load is 2x b128 per lane per tile.
// ---------------------------------------------------------------------------
__global__ void pack_w(const float* __restrict__ src, _Float16* __restrict__ dst,
                       int cols, int nTiles, int kTiles) {
  int d = blockIdx.x * blockDim.x + threadIdx.x;
  int total = nTiles * kTiles * 512;
  if (d >= total) return;
  int tile = d >> 9, w = d & 511, l = w >> 4, j = w & 15;
  int nt = tile / kTiles, kt = tile % kTiles;
  int row = nt * 16 + (l & 15);
  int col = kt * 32 + frag_k(l, j);
  dst[d] = (_Float16)src[(size_t)row * cols + col];
}

// ---------------------------------------------------------------------------
// vals[n][96]: encoder channel -> BPSK sequence fed to the GRU stack.
// ---------------------------------------------------------------------------
__global__ void make_vals(const float* __restrict__ x, _Float16* __restrict__ vals) {
  int i = blockIdx.x * blockDim.x + threadIdx.x;
  if (i >= 2048 * 96) return;
  int n = i / 96, s = i % 96;
  float v;
  if (s >= 88) {
    v = 1.f;
  } else {
    int r = s % 11;
    if (r < 9)      v = (x[n * 72 + (s / 11) * 9 + r] > 0.f) ? -1.f : 1.f;
    else if (r == 9) v = 1.f;
    else             v = -1.f;
  }
  vals[i] = (_Float16)v;
}

__global__ void zero_red(float* __restrict__ red) {
  int t = threadIdx.x;
  if (t < 258) red[t] = 0.f;
}

// ---------------------------------------------------------------------------
// VQ entropy terms: per (n, group-of-8) softmax over 256 codewords.
// ---------------------------------------------------------------------------
__global__ void prep_entropy(const float* __restrict__ x, float* __restrict__ red) {
  __shared__ float sprob[256];
  __shared__ float sent[2];
  int tid = threadIdx.x;
  sprob[tid] = 0.f;
  if (tid < 2) sent[tid] = 0.f;
  __syncthreads();

  int idx = blockIdx.x * 256 + tid;          // 72 blocks * 256 = 18432 exactly
  int n = idx / 9, k = idx % 9;
  float xv[8];
  float sabs = 0.f, msep = 0.f;
#pragma unroll
  for (int d = 0; d < 8; ++d) {
    float v = x[n * 72 + k * 8 + d];
    xv[d] = v;
    sabs += fabsf(v);
    float q = v > 0.f ? 1.f : -1.f;
    float dd = v - q;
    msep += dd * dd;
  }
  float mx = 200.f * sabs;                   // max possible logit
  float Z = 0.f, S = 0.f;
  for (int c = 0; c < 256; ++c) {
    float dot = 0.f;
#pragma unroll
    for (int d = 0; d < 8; ++d) dot += ((c >> d) & 1) ? xv[d] : -xv[d];
    float l = 200.f * dot - mx;
    float e = __expf(l);
    Z += e; S += e * l;
  }
  float invZ = 1.f / Z;
  for (int c = 0; c < 256; ++c) {
    float dot = 0.f;
#pragma unroll
    for (int d = 0; d < 8; ++d) dot += ((c >> d) & 1) ? xv[d] : -xv[d];
    atomicAdd(&sprob[c], __expf(200.f * dot - mx) * invZ);
  }
  atomicAdd(&sent[0], logf(Z) - S * invZ);
  atomicAdd(&sent[1], msep);
  __syncthreads();
  atomicAdd(&red[tid], sprob[tid]);
  if (tid == 0) { atomicAdd(&red[256], sent[0]); atomicAdd(&red[257], sent[1]); }
}

__global__ void finalize_aux(const float* __restrict__ red, float* __restrict__ out_aux) {
  __shared__ float s[256];
  int t = threadIdx.x;
  float p = red[t] * (1.f / 18432.f);
  s[t] = -(p * logf(fmaxf(p, 1e-9f)));
  __syncthreads();
  for (int o = 128; o; o >>= 1) { if (t < o) s[t] += s[t + o]; __syncthreads(); }
  if (t == 0) {
    float cbe = s[0];
    float pse = red[256] * (1.f / 18432.f);
    float mse = red[257] * (1.f / (2048.f * 72.f));
    out_aux[0] = 0.1f * (pse - cbe) + 0.25f * mse;
  }
}

// ---------------------------------------------------------------------------
// One bidirectional GRU layer. grid = (128 batch tiles, 2 directions),
// block = 256 (8 waves). Wave w owns gate columns [16w,16w+16).
// h lives in LDS; X_t tiles (layers 1/2) double-buffered in LDS via TDM.
// ---------------------------------------------------------------------------
__launch_bounds__(256)
__global__ void gru_layer(const _Float16* __restrict__ xin,   // layer0: vals[2048][96] else [96][2048][256]
                          _Float16* __restrict__ hout,        // [96][2048][256]
                          const _Float16* __restrict__ wih_f, const _Float16* __restrict__ whh_f,
                          const float* __restrict__ bih_f,    const float* __restrict__ bhh_f,
                          const _Float16* __restrict__ wih_b, const _Float16* __restrict__ whh_b,
                          const float* __restrict__ bih_b,    const float* __restrict__ bhh_b,
                          int layer0, int kTin) {
  __shared__ __align__(32) _Float16 hls[16 * 128];      // h state (f16)
  __shared__ __align__(32) _Float16 vls[16 * 96];       // layer0 vals tile
  __shared__ __align__(32) _Float16 xls[2][16 * 256];   // TDM double buffer

  const int dir  = blockIdx.y;
  const int n0   = blockIdx.x * 16;
  const int tid  = threadIdx.x;
  const int lane = tid & 31;
  const int wv   = tid >> 5;            // wave id 0..7
  const int mrow = lane & 15;
  const int sel  = lane >> 4;

  const _Float16* __restrict__ wih = dir ? wih_b : wih_f;
  const _Float16* __restrict__ whh = dir ? whh_b : whh_f;
  const float* __restrict__ bih    = dir ? bih_b : bih_f;
  const float* __restrict__ bhh    = dir ? bhh_b : bhh_f;

  for (int i = tid; i < 16 * 128; i += 256) hls[i] = (_Float16)0.f;
  if (layer0) {
    for (int i = tid; i < 16 * 96; i += 256)
      vls[i] = xin[(size_t)(n0 + i / 96) * 96 + (i % 96)];
  } else if (wv == 0) {
    // prefetch X tile for step 0 via Tensor Data Mover
    const int tt0 = dir ? 95 : 0;
    tdm_load_1d((unsigned int)(uintptr_t)&xls[0][0],
                xin + ((size_t)tt0 * 2048 + n0) * 256, 8192);
  }
  __syncthreads();

  const int col = wv * 16 + mrow;       // this thread's gate column (0..127)
  const float bi_r = bih[col], bi_z = bih[col + 128], bi_n = bih[col + 256];
  const float bh_r = bhh[col], bh_z = bhh[col + 128], bh_n = bhh[col + 256];

  for (int step = 0; step < 96; ++step) {
    const int tt = dir ? (95 - step) : step;
    const int buf = step & 1;

    if (!layer0) {
      // issue TDM for step+1 into the other buffer, then wait for current
      if (wv == 0) {
        if (step + 1 < 96) {
          const int tt1 = dir ? (95 - (step + 1)) : (step + 1);
          tdm_load_1d((unsigned int)(uintptr_t)&xls[buf ^ 1][0],
                      xin + ((size_t)tt1 * 2048 + n0) * 256, 8192);
          __builtin_amdgcn_s_wait_tensorcnt(1);   // current tile done
        } else {
          __builtin_amdgcn_s_wait_tensorcnt(0);
        }
      }
      __syncthreads();                            // publish TDM data to WG
    }

    v8f ar = vzero(), az = vzero(), an = vzero();
    v8f hr = vzero(), hz = vzero(), hn = vzero();

    // ---- input GEMM: gi = X_t @ Wih^T ----
    for (int kt = 0; kt < kTin; ++kt) {
      V16 a;
      if (layer0) {
        const int kb = kt * 32;
        if (kb >= tt) break;                       // fully masked tiles: skip
        const _Float16* r = &vls[mrow * 96 + kb];
        a.h[0] = *(const v8h*)(r + sel * 8);
        a.h[1] = *(const v8h*)(r + 16 + sel * 8);
        if (tt < kb + 32) {                        // boundary tile: mask s>=t
#pragma unroll
          for (int j = 0; j < 16; ++j) {
            int k = kb + sel * 8 + j + (j & 8);
            if (k >= tt) a.e[j] = (_Float16)0.f;
          }
        }
      } else {
        const _Float16* base = &xls[buf][mrow * 256 + kt * 32];
        a.h[0] = *(const v8h*)(base + sel * 8);
        a.h[1] = *(const v8h*)(base + 16 + sel * 8);
      }
      V16 b;
      const size_t t_r = ((size_t)wv * kTin + kt) * 512;
      __builtin_prefetch(wih + t_r + 512, 0, 1);   // global_prefetch_b8
      b.h[0] = *(const v8h*)(wih + t_r + lane * 16);
      b.h[1] = *(const v8h*)(wih + t_r + lane * 16 + 8);
      ar = wmma_f16(a.v, b.v, ar);
      const size_t t_z = ((size_t)(8 + wv) * kTin + kt) * 512;
      b.h[0] = *(const v8h*)(wih + t_z + lane * 16);
      b.h[1] = *(const v8h*)(wih + t_z + lane * 16 + 8);
      az = wmma_f16(a.v, b.v, az);
      const size_t t_n = ((size_t)(16 + wv) * kTin + kt) * 512;
      b.h[0] = *(const v8h*)(wih + t_n + lane * 16);
      b.h[1] = *(const v8h*)(wih + t_n + lane * 16 + 8);
      an = wmma_f16(a.v, b.v, an);
    }

    // ---- recurrent GEMM: gh = h @ Whh^T (K = 128) ----
#pragma unroll
    for (int kt = 0; kt < 4; ++kt) {
      V16 a;
      const _Float16* base = &hls[mrow * 128 + kt * 32];
      a.h[0] = *(const v8h*)(base + sel * 8);
      a.h[1] = *(const v8h*)(base + 16 + sel * 8);
      V16 b;
      const size_t t_r = ((size_t)wv * 4 + kt) * 512;
      b.h[0] = *(const v8h*)(whh + t_r + lane * 16);
      b.h[1] = *(const v8h*)(whh + t_r + lane * 16 + 8);
      hr = wmma_f16(a.v, b.v, hr);
      const size_t t_z = ((size_t)(8 + wv) * 4 + kt) * 512;
      b.h[0] = *(const v8h*)(whh + t_z + lane * 16);
      b.h[1] = *(const v8h*)(whh + t_z + lane * 16 + 8);
      hz = wmma_f16(a.v, b.v, hz);
      const size_t t_n = ((size_t)(16 + wv) * 4 + kt) * 512;
      b.h[0] = *(const v8h*)(whh + t_n + lane * 16);
      b.h[1] = *(const v8h*)(whh + t_n + lane * 16 + 8);
      hn = wmma_f16(a.v, b.v, hn);
    }

    __syncthreads();   // all reads of hls complete before owners update it

    // ---- gating: each thread owns (rows sel*8..sel*8+7, column col) ----
#pragma unroll
    for (int r = 0; r < 8; ++r) {
      const int m = sel * 8 + r;
      float gr = (ar[r] + bi_r) + (hr[r] + bh_r);
      float gz = (az[r] + bi_z) + (hz[r] + bh_z);
      float rg = 1.f / (1.f + __expf(-gr));
      float zg = 1.f / (1.f + __expf(-gz));
      float ng = tanhf((an[r] + bi_n) + rg * (hn[r] + bh_n));
      float hp = (float)hls[m * 128 + col];
      float hv = (1.f - zg) * ng + zg * hp;
      hls[m * 128 + col] = (_Float16)hv;
      hout[((size_t)tt * 2048 + n0 + m) * 256 + dir * 128 + col] = (_Float16)hv;
    }
    __syncthreads();   // h_new visible before next step's fragment loads
  }
}

// ---------------------------------------------------------------------------
// MLP head + hard decision. grid = (128 batch tiles, 96 positions).
// Feature tile staged by TDM; 256->128->32->1, hard sign output.
// ---------------------------------------------------------------------------
__launch_bounds__(256)
__global__ void mlp_head(const _Float16* __restrict__ feat,  // [96][2048][256]
                         const _Float16* __restrict__ w1p, const float* __restrict__ b1,
                         const _Float16* __restrict__ w2p, const float* __restrict__ b2,
                         const float* __restrict__ w3,     const float* __restrict__ b3,
                         float* __restrict__ out) {
  __shared__ __align__(32) _Float16 fls[16 * 256];
  __shared__ __align__(32) _Float16 z1[16 * 128];
  __shared__ float z2[16 * 32];

  const int s    = blockIdx.y;
  const int n0   = blockIdx.x * 16;
  const int tid  = threadIdx.x;
  const int lane = tid & 31;
  const int wv   = tid >> 5;
  const int mrow = lane & 15;
  const int sel  = lane >> 4;

  if (wv == 0) {
    tdm_load_1d((unsigned int)(uintptr_t)fls,
                feat + ((size_t)s * 2048 + n0) * 256, 8192);
    __builtin_amdgcn_s_wait_tensorcnt(0);
  }
  __syncthreads();

  // gemm1: wave wv -> output cols [16wv, 16wv+16), K = 256
  v8f acc = vzero();
#pragma unroll
  for (int kt = 0; kt < 8; ++kt) {
    V16 a;
    const _Float16* base = &fls[mrow * 256 + kt * 32];
    a.h[0] = *(const v8h*)(base + sel * 8);
    a.h[1] = *(const v8h*)(base + 16 + sel * 8);
    V16 b;
    const size_t tb = ((size_t)wv * 8 + kt) * 512;
    b.h[0] = *(const v8h*)(w1p + tb + lane * 16);
    b.h[1] = *(const v8h*)(w1p + tb + lane * 16 + 8);
    acc = wmma_f16(a.v, b.v, acc);
  }
  {
    const int col = wv * 16 + mrow;
    const float bb = b1[col];
#pragma unroll
    for (int r = 0; r < 8; ++r) {
      float v = acc[r] + bb;
      z1[(sel * 8 + r) * 128 + col] = (_Float16)(v > 0.f ? v : 0.f);
    }
  }
  __syncthreads();

  // gemm2: waves 0,1 -> 32 outputs, K = 128
  if (wv < 2) {
    v8f a2 = vzero();
#pragma unroll
    for (int kt = 0; kt < 4; ++kt) {
      V16 a;
      const _Float16* base = &z1[mrow * 128 + kt * 32];
      a.h[0] = *(const v8h*)(base + sel * 8);
      a.h[1] = *(const v8h*)(base + 16 + sel * 8);
      V16 b;
      const size_t tb = ((size_t)wv * 4 + kt) * 512;
      b.h[0] = *(const v8h*)(w2p + tb + lane * 16);
      b.h[1] = *(const v8h*)(w2p + tb + lane * 16 + 8);
      a2 = wmma_f16(a.v, b.v, a2);
    }
    const int c2 = wv * 16 + mrow;
    const float bb2 = b2[c2];
#pragma unroll
    for (int r = 0; r < 8; ++r) {
      float v = a2[r] + bb2;
      z2[(sel * 8 + r) * 32 + c2] = v > 0.f ? v : 0.f;
    }
  }
  __syncthreads();

  // head: 32->1 dot + hard decision; only data-carrying positions emit
  const int rr = s % 11;
  if (s < 88 && rr < 9 && wv == 0 && lane < 16) {
    float a3 = b3[0];
#pragma unroll
    for (int k = 0; k < 32; ++k) a3 += z2[lane * 32 + k] * w3[k];
    const float hard = a3 > 0.f ? 1.f : -1.f;   // sigmoid(a3) > 0.5 <=> a3 > 0
    const int c = (s / 11) * 9 + rr;
    const size_t n = (size_t)(n0 + lane);
    out[n * 72 + c] = hard;                      // final
    out[147457 + n * 72 + c] = hard;             // outt_2
  }
}

// ---------------------------------------------------------------------------
extern "C" void kernel_launch(void* const* d_in, const int* in_sizes, int n_in,
                              void* d_out, int out_size, void* d_ws, size_t ws_size,
                              hipStream_t stream) {
  (void)in_sizes; (void)n_in; (void)out_size; (void)ws_size;
  const float* x = (const float*)d_in[0];
  const float* rW[3][2][4];   // [layer][dir][Wih,Whh,bih,bhh]
  for (int l = 0; l < 3; ++l)
    for (int d = 0; d < 2; ++d)
      for (int p = 0; p < 4; ++p)
        rW[l][d][p] = (const float*)d_in[3 + l * 8 + d * 4 + p];
  const float* W1 = (const float*)d_in[27];
  const float* b1 = (const float*)d_in[28];
  const float* W2 = (const float*)d_in[29];
  const float* b2 = (const float*)d_in[30];
  const float* W3 = (const float*)d_in[31];
  const float* b3 = (const float*)d_in[32];

  char* p = (char*)d_ws;
  auto carve = [&](size_t bytes) -> char* {
    char* r = p;
    p += (bytes + 255) & ~(size_t)255;
    return r;
  };
  _Float16* vals = (_Float16*)carve((size_t)2048 * 96 * 2);
  _Float16* bufA = (_Float16*)carve((size_t)96 * 2048 * 256 * 2);
  _Float16* bufB = (_Float16*)carve((size_t)96 * 2048 * 256 * 2);
  _Float16* wih_p[3][2];
  _Float16* whh_p[3][2];
  for (int l = 0; l < 3; ++l)
    for (int d = 0; d < 2; ++d) {
      const int kT = l ? 8 : 3;
      wih_p[l][d] = (_Float16*)carve((size_t)24 * kT * 512 * 2);
      whh_p[l][d] = (_Float16*)carve((size_t)24 * 4 * 512 * 2);
    }
  _Float16* w1p = (_Float16*)carve((size_t)8 * 8 * 512 * 2);
  _Float16* w2p = (_Float16*)carve((size_t)2 * 4 * 512 * 2);
  float* red = (float*)carve(2048);

  zero_red<<<1, 258, 0, stream>>>(red);
  make_vals<<<768, 256, 0, stream>>>(x, vals);
  prep_entropy<<<72, 256, 0, stream>>>(x, red);

  for (int l = 0; l < 3; ++l)
    for (int d = 0; d < 2; ++d) {
      const int kin = l ? 256 : 96;
      const int kT = kin / 32;
      pack_w<<<(24 * kT * 512 + 255) / 256, 256, 0, stream>>>(rW[l][d][0], wih_p[l][d], kin, 24, kT);
      pack_w<<<(24 * 4 * 512 + 255) / 256, 256, 0, stream>>>(rW[l][d][1], whh_p[l][d], 128, 24, 4);
    }
  pack_w<<<(8 * 8 * 512 + 255) / 256, 256, 0, stream>>>(W1, w1p, 256, 8, 8);
  pack_w<<<(2 * 4 * 512 + 255) / 256, 256, 0, stream>>>(W2, w2p, 128, 2, 4);

  dim3 gg(128, 2);
  gru_layer<<<gg, 256, 0, stream>>>(vals, bufA,
      wih_p[0][0], whh_p[0][0], rW[0][0][2], rW[0][0][3],
      wih_p[0][1], whh_p[0][1], rW[0][1][2], rW[0][1][3], 1, 3);
  gru_layer<<<gg, 256, 0, stream>>>(bufA, bufB,
      wih_p[1][0], whh_p[1][0], rW[1][0][2], rW[1][0][3],
      wih_p[1][1], whh_p[1][1], rW[1][1][2], rW[1][1][3], 0, 8);
  gru_layer<<<gg, 256, 0, stream>>>(bufB, bufA,
      wih_p[2][0], whh_p[2][0], rW[2][0][2], rW[2][0][3],
      wih_p[2][1], whh_p[2][1], rW[2][1][2], rW[2][1][3], 0, 8);

  dim3 gm(128, 96);
  mlp_head<<<gm, 256, 0, stream>>>(bufA, w1p, b1, w2p, b2, W3, b3, (float*)d_out);
  finalize_aux<<<1, 256, 0, stream>>>(red, (float*)d_out + 147456);
}
